// MLPExperts_76965813944408
// MI455X (gfx1250) — matrix operand order repaired and checked by reading, not compiled, not run on hardware
//
#include <hip/hip_runtime.h>
#include <cstdint>

// ---------------------------------------------------------------------------
// MLPExperts for MI455X (gfx1250): f16 WMMA GEMMs with fp32 accumulation,
// double-buffered LDS tiles fed by GLOBAL_LOAD_ASYNC_TO_LDS_B128.
//   h1 = leaky(x @ w1); h2 = leaky(h1 @ w2); out = h2 @ w3
// E=8, N=2048, D=1024, H=4096.  All GEMM dims divisible by 128/64.
// ---------------------------------------------------------------------------

typedef __attribute__((ext_vector_type(16))) _Float16 v16h;
typedef __attribute__((ext_vector_type(8)))  _Float16 v8h;
typedef __attribute__((ext_vector_type(8)))  float    v8f;

static constexpr int   E_  = 8;
static constexpr int   NT_ = 2048;   // tokens per expert
static constexpr int   D_  = 1024;
static constexpr int   H_  = 4096;
static constexpr float NEG_SLOPE = 0.01f;

#if defined(__HIP_DEVICE_COMPILE__) && defined(__gfx1250__) && \
    __has_builtin(__builtin_amdgcn_global_load_async_to_lds_b128)
#define USE_ASYNC 1
// Exact parameter types per hipcc diagnostic: param 1 is
// '__attribute__((__vector_size__(4 * sizeof(int)))) int __device__ *'
// (= int4 in addrspace(1)); param 2 analogously int4 in addrspace(3) (LDS).
typedef int v4i_ __attribute__((vector_size(16)));
typedef __attribute__((address_space(1))) v4i_ GlbV4;
typedef __attribute__((address_space(3))) v4i_ LdsV4;
#else
#define USE_ASYNC 0
#endif

__device__ __forceinline__ void wait_async_zero() {
#if USE_ASYNC
#if __has_builtin(__builtin_amdgcn_s_wait_asynccnt)
  __builtin_amdgcn_s_wait_asynccnt(0);
#else
  asm volatile("s_wait_asynccnt 0x0" ::: "memory");
#endif
#endif
}

// ---------------------------------------------------------------------------
// fp32 -> f16 straight conversion (for x). 8 elements / thread.
// ---------------------------------------------------------------------------
__global__ __launch_bounds__(256) void cvt_f32_f16(const float* __restrict__ in,
                                                   _Float16* __restrict__ out,
                                                   size_t n) {
  size_t i = ((size_t)blockIdx.x * blockDim.x + threadIdx.x) * 8;
  if (i + 8 <= n) {
    const float4 a = *reinterpret_cast<const float4*>(in + i);
    const float4 b = *reinterpret_cast<const float4*>(in + i + 4);
    v8h o;
    o[0] = (_Float16)a.x; o[1] = (_Float16)a.y;
    o[2] = (_Float16)a.z; o[3] = (_Float16)a.w;
    o[4] = (_Float16)b.x; o[5] = (_Float16)b.y;
    o[6] = (_Float16)b.z; o[7] = (_Float16)b.w;
    *reinterpret_cast<v8h*>(out + i) = o;
  }
}

// ---------------------------------------------------------------------------
// Transpose + convert weights: in [E][K][N] fp32 -> out [E][N][K] f16.
// Tiled 32x32 through LDS so both global sides are coalesced.
// Block: (32, 8). Grid: (N/32, K/32, E).
// ---------------------------------------------------------------------------
__global__ __launch_bounds__(256) void transpose_cvt(const float* __restrict__ in,
                                                     _Float16* __restrict__ out,
                                                     int K, int N) {
  __shared__ float tile[32][33];
  const int e = blockIdx.z;
  in  += (size_t)e * K * N;
  out += (size_t)e * N * K;
  const int n0 = blockIdx.x * 32;
  const int k0 = blockIdx.y * 32;
  const int tx = threadIdx.x;   // 0..31
  const int ty = threadIdx.y;   // 0..7
#pragma unroll
  for (int i = 0; i < 32; i += 8)
    tile[ty + i][tx] = in[(size_t)(k0 + ty + i) * N + (n0 + tx)];
  __syncthreads();
#pragma unroll
  for (int i = 0; i < 32; i += 8)
    out[(size_t)(n0 + ty + i) * K + (k0 + tx)] = (_Float16)tile[tx][ty + i];
}

// ---------------------------------------------------------------------------
// WMMA GEMM:  C[e] = A[e] (MxK, row major) * Bt[e]^T (Bt is NCxK, row major)
// Tile: 128x128x64, double buffered.  8 waves: wave (wm,wn) computes 32x64.
// ACT_F16=true : C is f16, leaky-relu applied.   false: C is f32 plain.
// ---------------------------------------------------------------------------
template <bool ACT_F16>
__global__ __launch_bounds__(256) void gemm_wmma(const _Float16* __restrict__ A,
                                                 const _Float16* __restrict__ Bt,
                                                 void* __restrict__ C,
                                                 int M, int K, int NC) {
  constexpr int BM = 128, BN = 128, BK = 64, LDST = 72;  // 72-half stride: conflict-free
  __shared__ __align__(16) _Float16 sA[2][BM][LDST];
  __shared__ __align__(16) _Float16 sB[2][BN][LDST];

  const int e = blockIdx.z;
  const _Float16* Ae = A  + (size_t)e * M * K;
  const _Float16* Be = Bt + (size_t)e * NC * K;

  const int tile_n = blockIdx.x * BN;
  const int tile_m = blockIdx.y * BM;

  const int tid  = threadIdx.x;
  const int lane = tid & 31;
  const int wave = tid >> 5;
  const int wm   = wave & 3;    // row strip: rows [32*wm, 32*wm+32)
  const int wn   = wave >> 2;   // col strip: cols [64*wn, 64*wn+64)
  const int lr   = lane & 15;
  const int hi   = lane >> 4;   // lane half-group (ISA VGPR layout)

  // Issue the 128x64 A tile and 128x64 Bt tile for buffer `buf` at k-offset k0.
  // 1024 chunks of 8 halves per matrix; 4 chunks per thread per matrix.
  auto issue_tile = [&](int buf, int k0) {
#pragma unroll
    for (int c = 0; c < 4; ++c) {
      const int idx = tid + c * 256;     // 0..1023
      const int row = idx >> 3;
      const int col = (idx & 7) * 8;
      const _Float16* ga = Ae + (size_t)(tile_m + row) * K + k0 + col;
      const _Float16* gb = Be + (size_t)(tile_n + row) * K + k0 + col;
      _Float16* la = &sA[buf][row][col];
      _Float16* lb = &sB[buf][row][col];
#if USE_ASYNC
      __builtin_amdgcn_global_load_async_to_lds_b128(
          (GlbV4*)(uintptr_t)ga, (LdsV4*)(uint32_t)(uintptr_t)la, 0, 0);
      __builtin_amdgcn_global_load_async_to_lds_b128(
          (GlbV4*)(uintptr_t)gb, (LdsV4*)(uint32_t)(uintptr_t)lb, 0, 0);
#else
      *reinterpret_cast<v8h*>(la) = *reinterpret_cast<const v8h*>(ga);
      *reinterpret_cast<v8h*>(lb) = *reinterpret_cast<const v8h*>(gb);
      if (c == 0) {
        __builtin_prefetch(ga + BK, 0, 1);
        __builtin_prefetch(gb + BK, 0, 1);
      }
#endif
    }
  };

  v8f acc[2][4];
#pragma unroll
  for (int s = 0; s < 2; ++s)
#pragma unroll
    for (int t = 0; t < 4; ++t)
      acc[s][t] = v8f{};

  const int ntiles = K / BK;
  issue_tile(0, 0);

  for (int it = 0; it < ntiles; ++it) {
    const int buf = it & 1;
    wait_async_zero();       // my tile-`it` loads have landed in LDS
    __syncthreads();         // everyone's landed; everyone done reading buf^1
    if (it + 1 < ntiles) issue_tile(buf ^ 1, (it + 1) * BK);

    // Two K=32 substeps over the staged 64-wide tile.
#pragma unroll
    for (int kk = 0; kk < 2; ++kk) {
      const int ko = kk * 32;

      // A frags: half h of lane holds K = (h<8 ? h : h+8) + 8*hi
      // => two contiguous 16B reads at ko + 8*hi and ko + 16 + 8*hi.
      v16h af[2];
#pragma unroll
      for (int s = 0; s < 2; ++s) {
        const int r = 32 * wm + 16 * s + lr;
        const v8h lo = *reinterpret_cast<const v8h*>(&sA[buf][r][ko + 8 * hi]);
        const v8h hs = *reinterpret_cast<const v8h*>(&sA[buf][r][ko + 16 + 8 * hi]);
        v16h a;
#pragma unroll
        for (int i = 0; i < 8; ++i) { a[i] = lo[i]; a[8 + i] = hs[i]; }
        af[s] = a;
      }

      // B frags (transposed tile): half h holds K = (h<8 ? h : h+8) + 8*hi
      // => two contiguous 16B reads per lane.  Load all 4 before any WMMA.
      v16h bf[4];
#pragma unroll
      for (int t = 0; t < 4; ++t) {
        const int n = 64 * wn + 16 * t + lr;
        const v8h lo = *reinterpret_cast<const v8h*>(&sB[buf][n][ko + 8 * hi]);
        const v8h hs = *reinterpret_cast<const v8h*>(&sB[buf][n][ko + 16 + 8 * hi]);
        v16h b;
#pragma unroll
        for (int i = 0; i < 8; ++i) { b[i] = lo[i]; b[8 + i] = hs[i]; }
        bf[t] = b;
      }

      // 8 back-to-back WMMAs.
#pragma unroll
      for (int t = 0; t < 4; ++t)
        acc[0][t] = __builtin_amdgcn_wmma_f32_16x16x32_f16(
            false, af[0], false, bf[t], (short)0, acc[0][t], false, false);
#pragma unroll
      for (int t = 0; t < 4; ++t)
        acc[1][t] = __builtin_amdgcn_wmma_f32_16x16x32_f16(
            false, af[1], false, bf[t], (short)0, acc[1][t], false, false);
    }
  }

  // ---- epilogue: C element i of v8f sits at row = i + 8*hi, col = lane&15 ----
  const size_t cbase = (size_t)e * M * NC;
#pragma unroll
  for (int s = 0; s < 2; ++s) {
    const int row0 = tile_m + 32 * wm + 16 * s + 8 * hi;
#pragma unroll
    for (int t = 0; t < 4; ++t) {
      const int col = tile_n + 64 * wn + 16 * t + lr;
#pragma unroll
      for (int i = 0; i < 8; ++i) {
        float v = acc[s][t][i];
        if (ACT_F16) {
          v = (v >= 0.0f) ? v : NEG_SLOPE * v;
          reinterpret_cast<_Float16*>(C)[cbase + (size_t)(row0 + i) * NC + col] =
              (_Float16)v;
        } else {
          reinterpret_cast<float*>(C)[cbase + (size_t)(row0 + i) * NC + col] = v;
        }
      }
    }
  }
}

// ---------------------------------------------------------------------------
// Launch: convert/transpose once per call, then 3 GEMMs (leaky fused into the first two).
// Workspace (f16 elements): xh 16M | w1t 32M | w2t 128M | w3t 32M | h1 64M | h2 64M
// = 672 MB total.
// ---------------------------------------------------------------------------
extern "C" void kernel_launch(void* const* d_in, const int* in_sizes, int n_in,
                              void* d_out, int out_size, void* d_ws, size_t ws_size,
                              hipStream_t stream) {
  const float* x  = (const float*)d_in[0];
  const float* w1 = (const float*)d_in[1];
  const float* w2 = (const float*)d_in[2];
  const float* w3 = (const float*)d_in[3];

  const size_t sz_x  = (size_t)E_ * NT_ * D_;   // 16M
  const size_t sz_w1 = (size_t)E_ * D_ * H_;    // 32M
  const size_t sz_w2 = (size_t)E_ * H_ * H_;    // 128M
  const size_t sz_w3 = (size_t)E_ * H_ * D_;    // 32M
  const size_t sz_h  = (size_t)E_ * NT_ * H_;   // 64M

  _Float16* xh  = (_Float16*)d_ws;
  _Float16* w1t = xh  + sz_x;
  _Float16* w2t = w1t + sz_w1;
  _Float16* w3t = w2t + sz_w2;
  _Float16* h1  = w3t + sz_w3;
  _Float16* h2  = h1  + sz_h;

  // fp32 -> f16 (x) ; fp32 -> f16 + transpose (weights -> [E][N][K])
  cvt_f32_f16<<<dim3((unsigned)(sz_x / 2048)), 256, 0, stream>>>(x, xh, sz_x);
  transpose_cvt<<<dim3(H_ / 32, D_ / 32, E_), dim3(32, 8), 0, stream>>>(w1, w1t, D_, H_);
  transpose_cvt<<<dim3(H_ / 32, H_ / 32, E_), dim3(32, 8), 0, stream>>>(w2, w2t, H_, H_);
  transpose_cvt<<<dim3(D_ / 32, H_ / 32, E_), dim3(32, 8), 0, stream>>>(w3, w3t, H_, D_);

  // layer 1: [2048x1024] @ [1024x4096] -> h1 (leaky, f16)
  gemm_wmma<true><<<dim3(H_ / 128, NT_ / 128, E_), 256, 0, stream>>>(
      xh, w1t, h1, NT_, D_, H_);
  // layer 2: [2048x4096] @ [4096x4096] -> h2 (leaky, f16)
  gemm_wmma<true><<<dim3(H_ / 128, NT_ / 128, E_), 256, 0, stream>>>(
      h1, w2t, h2, NT_, H_, H_);
  // layer 3: [2048x4096] @ [4096x1024] -> out (fp32)
  gemm_wmma<false><<<dim3(D_ / 128, NT_ / 128, E_), 256, 0, stream>>>(
      h2, w3t, d_out, NT_, H_, D_);
}
